// PKGN_60378650247741
// MI455X (gfx1250) — matrix operand order repaired and checked by reading, not compiled, not run on hardware
//
#include <hip/hip_runtime.h>
#include <hip/hip_bf16.h>
#include <stdint.h>

#define NNODES 32768
#define HDIM   768
#define NHEADS 2
#define NEDGE  131072
#define NLAB   4
#define ETOT   (NEDGE + NNODES)   /* 163840 */
#define NOUT   (NHEADS * HDIM)    /* 1536 */
#define NEG_SLOPE 0.2f

typedef __attribute__((ext_vector_type(16))) __bf16          v16bf;
typedef __attribute__((ext_vector_type(8)))  float           v8f;
typedef __attribute__((ext_vector_type(8)))  unsigned short  v8u;
typedef __attribute__((ext_vector_type(16))) unsigned short  v16u;
typedef __attribute__((ext_vector_type(4)))  float           v4f;
typedef __attribute__((ext_vector_type(4)))  unsigned short  v4u;

__device__ __forceinline__ unsigned short f2b(float f) {
  unsigned u = __float_as_uint(f);
  u += 0x7fffu + ((u >> 16) & 1u);          // round-to-nearest-even
  return (unsigned short)(u >> 16);
}
__device__ __forceinline__ float b2f(unsigned short b) {
  return __uint_as_float(((unsigned)b) << 16);
}
// monotone float <-> ordered-uint key (for atomicMax segment-max)
__device__ __forceinline__ unsigned f2ord(float f) {
  unsigned u = __float_as_uint(f);
  return (u & 0x80000000u) ? ~u : (u | 0x80000000u);
}
__device__ __forceinline__ float ord2f(unsigned o) {
  return (o & 0x80000000u) ? __uint_as_float(o & 0x7fffffffu)
                           : __uint_as_float(~o);
}

// gfx1250 async global->LDS copy (16B per lane), tracked by ASYNCcnt.
// vdst = VGPR with LDS byte address; vaddr = 64-bit global address.
__device__ __forceinline__ void async_cp16(unsigned lds_byte_addr, const void* g) {
  unsigned long long ga = (unsigned long long)(uintptr_t)g;
  asm volatile("global_load_async_to_lds_b128 %0, %1, off"
               :: "v"(lds_byte_addr), "v"(ga) : "memory");
}
__device__ __forceinline__ void wait_asynccnt0() {
  asm volatile("s_wait_asynccnt 0x0" ::: "memory");
}

// ---------------- zero / convert kernels ----------------

__global__ void k_zero4(v4f* __restrict__ p, long n4) {
  long i = (long)blockIdx.x * blockDim.x + threadIdx.x;
  v4f z; z[0] = z[1] = z[2] = z[3] = 0.0f;
  for (; i < n4; i += (long)gridDim.x * blockDim.x) p[i] = z;
}

__global__ void k_cvt_emb(const float* __restrict__ in, unsigned short* __restrict__ out) {
  size_t i = ((size_t)blockIdx.x * 256 + threadIdx.x) * 4;
  v4f f = *(const v4f*)(in + i);
  v4u b;
#pragma unroll
  for (int j = 0; j < 4; ++j) b[j] = f2b(f[j]);
  *(v4u*)(out + i) = b;
}

// W [H, NOUT] f32  ->  WT [NOUT, H] bf16 (both W_l and W_r in one launch)
__global__ void k_cvt_wT(const float* __restrict__ Wl, const float* __restrict__ Wr,
                         unsigned short* __restrict__ wlt, unsigned short* __restrict__ wrt) {
  size_t gid = (size_t)blockIdx.x * 256 + threadIdx.x;
  const size_t tot = (size_t)HDIM * NOUT;
  int which = gid >= tot;
  size_t idx = gid - (which ? tot : 0);
  size_t k = idx / NOUT, n = idx % NOUT;
  const float* W = which ? Wr : Wl;
  unsigned short* T = which ? wrt : wlt;
  T[n * HDIM + k] = f2b(W[idx]);
}

// ---------------- WMMA GEMM: X = emb(bf16) @ W(bf16), f32 accum, bf16 out ----
// grid = (NOUT/128, NNODES/128, 2), block = 256 (8 waves)
// Double-buffered LDS fed by async global->LDS b128 copies; 1 barrier/K-stage.
#define LDSS 40            /* padded row stride in halves (80B) -> conflict-free b128 */
#define TILE_HALVES (128 * LDSS)
#define TILE_BYTES  (TILE_HALVES * 2)   /* 10240 B */
#define KSTAGES (HDIM / 32)             /* 24 */

__global__ __launch_bounds__(256)
void k_gat_gemm(const unsigned short* __restrict__ A,
                const unsigned short* __restrict__ BT0,
                const unsigned short* __restrict__ BT1,
                unsigned short* __restrict__ X0,
                unsigned short* __restrict__ X1) {
  // [buf0 A][buf1 A][buf0 B][buf1 B]
  __shared__ unsigned short SH[4 * TILE_HALVES];

  const unsigned short* BT = blockIdx.z ? BT1 : BT0;
  unsigned short*       X  = blockIdx.z ? X1  : X0;

  const int tid  = threadIdx.x;
  const int lane = tid & 31;
  const int wave = tid >> 5;
  const int wm   = wave & 3;     // 4 wave-rows  x 32
  const int wn   = wave >> 2;    // 2 wave-cols  x 64
  const int r    = lane & 15;
  const int hi   = lane >> 4;
  const size_t mBase = (size_t)blockIdx.y * 128;
  const size_t nBase = (size_t)blockIdx.x * 128;

  // staging geometry: 512 x 16B chunks per tile, 2 A + 2 B chunks per thread
  const int row0 = tid >> 2;          // 0..63
  const int row1 = row0 + 64;         // 64..127
  const int seg  = tid & 3;           // 16B segment within a 64B row
  const unsigned ldsBase = (unsigned)(uintptr_t)&SH[0];
  const unsigned aoff0 = (unsigned)((row0 * LDSS + seg * 8) * 2);
  const unsigned aoff1 = (unsigned)((row1 * LDSS + seg * 8) * 2);
  const unsigned short* gA0 = &A [(mBase + row0) * HDIM + seg * 8];
  const unsigned short* gA1 = &A [(mBase + row1) * HDIM + seg * 8];
  const unsigned short* gB0 = &BT[(nBase + row0) * HDIM + seg * 8];
  const unsigned short* gB1 = &BT[(nBase + row1) * HDIM + seg * 8];

  v8f acc[2][4];
#pragma unroll
  for (int s = 0; s < 2; ++s)
#pragma unroll
    for (int t = 0; t < 4; ++t)
#pragma unroll
      for (int i = 0; i < 8; ++i) acc[s][t][i] = 0.0f;

  // prologue: stage K-stage 0 into buffer 0
  {
    const unsigned ab = ldsBase;                       // A buf0
    const unsigned bb = ldsBase + 2u * TILE_BYTES;     // B buf0
    async_cp16(ab + aoff0, gA0);
    async_cp16(ab + aoff1, gA1);
    async_cp16(bb + aoff0, gB0);
    async_cp16(bb + aoff1, gB1);
    wait_asynccnt0();
    __syncthreads();
  }

  for (int ks = 0; ks < KSTAGES; ++ks) {
    const int cur = ks & 1;
    const int nxt = cur ^ 1;
    const bool more = (ks + 1) < KSTAGES;

    // kick off async fetch of next tile into the other buffer
    if (more) {
      const int k1 = (ks + 1) * 32;
      const unsigned ab = ldsBase + (unsigned)nxt * TILE_BYTES;
      const unsigned bb = ldsBase + (2u + (unsigned)nxt) * TILE_BYTES;
      async_cp16(ab + aoff0, gA0 + k1);
      async_cp16(ab + aoff1, gA1 + k1);
      async_cp16(bb + aoff0, gB0 + k1);
      async_cp16(bb + aoff1, gB1 + k1);
    }

    const unsigned short* As = &SH[cur * TILE_HALVES];
    const unsigned short* Bs = &SH[(2 + cur) * TILE_HALVES];

    // B fragments: col = r, halves 0..15 = K hi*16 + 0..15 (contiguous 32B)
    v16bf bfr[4];
#pragma unroll
    for (int t = 0; t < 4; ++t) {
      const unsigned short* bp = &Bs[(wn * 64 + t * 16 + r) * LDSS + hi * 16];
      v8u lo  = *(const v8u*)bp;
      v8u hi2 = *(const v8u*)(bp + 8);
      v16u u;
#pragma unroll
      for (int i = 0; i < 8; ++i) { u[i] = lo[i]; u[i + 8] = hi2[i]; }
      bfr[t] = __builtin_bit_cast(v16bf, u);
    }
    // A fragments: row = r, halves 0..15 = K hi*16 + 0..15 (contiguous 32B)
#pragma unroll
    for (int s = 0; s < 2; ++s) {
      const unsigned short* ap = &As[(wm * 32 + s * 16 + r) * LDSS + hi * 16];
      v8u lo  = *(const v8u*)ap;
      v8u hi2 = *(const v8u*)(ap + 8);
      v16u u;
#pragma unroll
      for (int i = 0; i < 8; ++i) { u[i] = lo[i]; u[i + 8] = hi2[i]; }
      v16bf afr = __builtin_bit_cast(v16bf, u);
#pragma unroll
      for (int t = 0; t < 4; ++t)
        acc[s][t] = __builtin_amdgcn_wmma_f32_16x16x32_bf16(
            false, afr, false, bfr[t], (short)0, acc[s][t], false, false);
    }

    if (more) {
      wait_asynccnt0();   // this wave's async copies into nxt have landed
      __syncthreads();    // everyone's copies visible, everyone done reading cur
    }
  }

  // epilogue: C layout -> row = hi*8+i, col = r
#pragma unroll
  for (int s = 0; s < 2; ++s)
#pragma unroll
    for (int t = 0; t < 4; ++t) {
      size_t col = nBase + wn * 64 + t * 16 + r;
#pragma unroll
      for (int i = 0; i < 8; ++i) {
        size_t row = mBase + wm * 32 + s * 16 + hi * 8 + i;
        X[row * NOUT + col] = f2b(acc[s][t][i]);
      }
    }
}

// ---------------- edge kernels (wave-per-edge) ----------------

__global__ __launch_bounds__(256)
void k_edge_logits(const unsigned short* __restrict__ xl,
                   const unsigned short* __restrict__ xr,
                   const int* __restrict__ edges,
                   const float* __restrict__ att,
                   float* __restrict__ elog,
                   unsigned* __restrict__ mOrd) {
  int wid  = (blockIdx.x * 256 + threadIdx.x) >> 5;
  int lane = threadIdx.x & 31;
  int s, d;
  if (wid < NEDGE) { s = edges[wid]; d = edges[NEDGE + wid]; }
  else             { s = wid - NEDGE; d = s; }
  size_t sb = (size_t)s * NOUT, db = (size_t)d * NOUT;
  float a0 = 0.f, a1 = 0.f;
  for (int j = lane; j < HDIM; j += 32) {
    float v0 = b2f(xl[sb + j]) + b2f(xr[db + j]);
    v0 = v0 > 0.f ? v0 : NEG_SLOPE * v0;
    a0 += v0 * att[j];
    float v1 = b2f(xl[sb + HDIM + j]) + b2f(xr[db + HDIM + j]);
    v1 = v1 > 0.f ? v1 : NEG_SLOPE * v1;
    a1 += v1 * att[HDIM + j];
  }
#pragma unroll
  for (int off = 16; off > 0; off >>= 1) {
    a0 += __shfl_xor(a0, off, 32);
    a1 += __shfl_xor(a1, off, 32);
  }
  if (lane == 0) {
    elog[(size_t)wid * 2]     = a0;
    elog[(size_t)wid * 2 + 1] = a1;
    atomicMax(&mOrd[d * 2],     f2ord(a0));
    atomicMax(&mOrd[d * 2 + 1], f2ord(a1));
  }
}

__global__ void k_edge_w(float* __restrict__ elog, const unsigned* __restrict__ mOrd,
                         const int* __restrict__ edges, float* __restrict__ denom) {
  int i = blockIdx.x * 256 + threadIdx.x;   // < ETOT*2
  int e = i >> 1, h = i & 1;
  int d = (e < NEDGE) ? edges[NEDGE + e] : (e - NEDGE);
  float m = ord2f(mOrd[d * 2 + h]);
  float w = __expf(elog[i] - m);
  elog[i] = w;
  atomicAdd(&denom[d * 2 + h], w);
}

__global__ __launch_bounds__(256)
void k_edge_agg(const unsigned short* __restrict__ xl,
                const float* __restrict__ wv,
                const float* __restrict__ denom,
                const int* __restrict__ edges,
                float* __restrict__ gat) {
  int wid  = (blockIdx.x * 256 + threadIdx.x) >> 5;
  int lane = threadIdx.x & 31;
  int s, d;
  if (wid < NEDGE) { s = edges[wid]; d = edges[NEDGE + wid]; }
  else             { s = wid - NEDGE; d = s; }
  float al0 = wv[(size_t)wid * 2]     / denom[d * 2];
  float al1 = wv[(size_t)wid * 2 + 1] / denom[d * 2 + 1];
  size_t sb = (size_t)s * NOUT;
  float* g = gat + (size_t)d * HDIM;
  for (int j = lane; j < HDIM; j += 32)
    atomicAdd(&g[j], 0.5f * (al0 * b2f(xl[sb + j]) + al1 * b2f(xl[sb + HDIM + j])));
}

// ---------------- pooling / classifier ----------------

__global__ void k_count(const int* __restrict__ mask, float* __restrict__ countf) {
  __shared__ int sh[256];
  int t = threadIdx.x, c = 0;
  for (int i = t; i < NNODES; i += 256) c += mask[i];
  sh[t] = c; __syncthreads();
  for (int s = 128; s > 0; s >>= 1) { if (t < s) sh[t] += sh[t + s]; __syncthreads(); }
  if (t == 0) *countf = (float)sh[0];
}

__global__ void k_pool(const float* __restrict__ gat, const int* __restrict__ mask,
                       float* __restrict__ pooled) {
  int t = threadIdx.x;
  int rowBase = blockIdx.x * 128;
  float s0 = 0.f, s1 = 0.f, s2 = 0.f;
  for (int r = 0; r < 128; ++r) {
    int row = rowBase + r;
    if (mask[row]) {
      const float* g = gat + (size_t)row * HDIM;
      s0 += g[t]; s1 += g[t + 256]; s2 += g[t + 512];
    }
  }
  atomicAdd(&pooled[t],       s0);
  atomicAdd(&pooled[t + 256], s1);
  atomicAdd(&pooled[t + 512], s2);
}

__global__ void k_final(const float* __restrict__ pooled, const float* __restrict__ countf,
                        const float* __restrict__ gbias, const float* __restrict__ clsW,
                        const float* __restrict__ clsb, const int* __restrict__ labels,
                        float* __restrict__ out) {
  __shared__ float lin[8];
  int t = threadIdx.x;
  float cnt = *countf;
  float inv = 1.0f / (cnt + 1e-8f);
  if (t < 8) {
    int l = t >> 1, o = t & 1;
    float s = clsb[l * 2 + o];
    for (int h = 0; h < HDIM; ++h) {
      float psy = (pooled[h] + cnt * gbias[h]) * inv;
      s += psy * clsW[(l * HDIM + h) * 2 + o];
    }
    lin[t] = s;
  }
  __syncthreads();
  if (t == 0) {
    float loss = 0.f, acc = 0.f;
    for (int l = 0; l < NLAB; ++l) {
      float a = lin[l * 2], b = lin[l * 2 + 1];
      float m = fmaxf(a, b);
      float ea = __expf(a - m), eb = __expf(b - m);
      float s = ea + eb;
      float p0 = ea / s, p1 = eb / s;
      // reference applies log_softmax to the probs (double softmax)
      float mm = fmaxf(p0, p1);
      float z = __expf(p0 - mm) + __expf(p1 - mm);
      float lp0 = p0 - mm - __logf(z), lp1 = p1 - mm - __logf(z);
      int lab = labels[l];
      loss += -(lab ? lp1 : lp0);
      int pred = (p1 > p0) ? 1 : 0;
      out[1 + l] = (float)pred;
      acc += (pred == lab) ? 1.f : 0.f;
    }
    out[0] = loss;
    out[5] = acc;
  }
}

// ---------------- host launch ----------------

extern "C" void kernel_launch(void* const* d_in, const int* in_sizes, int n_in,
                              void* d_out, int out_size, void* d_ws, size_t ws_size,
                              hipStream_t stream) {
  const float* emb    = (const float*)d_in[0];
  const int*   mask   = (const int*)  d_in[1];
  const int*   edges  = (const int*)  d_in[2];
  const int*   labels = (const int*)  d_in[3];
  const float* Wl     = (const float*)d_in[4];
  const float* Wr     = (const float*)d_in[5];
  const float* att    = (const float*)d_in[6];
  const float* gbias  = (const float*)d_in[7];
  const float* clsW   = (const float*)d_in[8];
  const float* clsb   = (const float*)d_in[9];
  float* out = (float*)d_out;

  char* ws = (char*)d_ws;
  size_t off = 0;
  unsigned short* embB = (unsigned short*)(ws + off); off += (size_t)NNODES * HDIM * 2;
  unsigned short* wlt  = (unsigned short*)(ws + off); off += (size_t)NOUT * HDIM * 2;
  unsigned short* wrt  = (unsigned short*)(ws + off); off += (size_t)NOUT * HDIM * 2;
  unsigned short* xl   = (unsigned short*)(ws + off); off += (size_t)NNODES * NOUT * 2;
  unsigned short* xr   = (unsigned short*)(ws + off); off += (size_t)NNODES * NOUT * 2;
  // ---- contiguous zeroed region: mOrd | denom | pooled | gat ----
  unsigned* mOrd = (unsigned*)(ws + off); off += (size_t)NNODES * NHEADS * 4;
  float* denom   = (float*)(ws + off);    off += (size_t)NNODES * NHEADS * 4;
  float* pooled  = (float*)(ws + off);    off += (size_t)HDIM * 4;
  float* gat     = (float*)(ws + off);    off += (size_t)NNODES * HDIM * 4;
  // ---- end zeroed region ----
  float* elog    = (float*)(ws + off);    off += (size_t)ETOT * NHEADS * 4;
  float* countf  = (float*)(ws + off);    off += 256;

  const long zeroFloats = (long)NNODES * NHEADS * 2 + HDIM + (long)NNODES * HDIM;
  k_zero4<<<4096, 256, 0, stream>>>((v4f*)mOrd, zeroFloats / 4);

  k_cvt_emb<<<(NNODES * HDIM / 4) / 256, 256, 0, stream>>>(emb, embB);
  k_cvt_wT<<<(2 * HDIM * NOUT) / 256, 256, 0, stream>>>(Wl, Wr, wlt, wrt);

  dim3 gGemm(NOUT / 128, NNODES / 128, 2);   // (12, 256, 2)
  k_gat_gemm<<<gGemm, 256, 0, stream>>>(embB, wlt, wrt, xl, xr);

  k_edge_logits<<<ETOT / 8, 256, 0, stream>>>(xl, xr, edges, att, elog, mOrd);
  k_edge_w<<<(ETOT * 2) / 256, 256, 0, stream>>>(elog, mOrd, edges, denom);
  k_edge_agg<<<ETOT / 8, 256, 0, stream>>>(xl, elog, denom, edges, gat);

  k_count<<<1, 256, 0, stream>>>(mask, countf);
  k_pool<<<NNODES / 128, 256, 0, stream>>>(gat, mask, pooled);
  k_final<<<1, 64, 0, stream>>>(pooled, countf, gbias, clsW, clsb, labels, out);
}
